// MetaController_41034117546195
// MI455X (gfx1250) — matrix operand order repaired and checked by reading, not compiled, not run on hardware
//
#include <hip/hip_runtime.h>
#include <hip/hip_bf16.h>

// CDNA5 (gfx1250) wave32 WMMA types
typedef __attribute__((ext_vector_type(16))) __bf16 v16bf;
typedef __attribute__((ext_vector_type(8)))  float  v8f;

static __device__ __forceinline__ float sigm(float x) { return 1.0f / (1.0f + __expf(-x)); }

// ---------------------------------------------------------------------------
// Pack a (K x N) f32 weight into zero-padded bf16 WMMA B-fragment layout,
// writing column-chunks [nbOff, nbOff+ceil(N/16)) of a fragment set that has
// NBtot chunks per k-step.  bT=0: src[k*ldb+c] ; bT=1: src[c*ldb+k].
// Fragment layout (dwords): frag[((ks*NBtot + nb)*32 + lane)*8 + v]
//  lane: col = lane&15, hf = lane>>4 ; dword v packs B[k], B[k+1],
//  k = ks*32 + hf*16 + 2v  (contiguous k order; A fragments use the same).
// ---------------------------------------------------------------------------
__global__ void k_pack_wfrag(const float* __restrict__ W, int ldb, int bT,
                             int K, int N, int NBl, int nbOff, int NBtot,
                             int total, unsigned* __restrict__ frag)
{
    int idx = blockIdx.x * 256 + threadIdx.x;
    if (idx >= total) return;
    int v    = idx & 7;
    int lane = (idx >> 3) & 31;
    int fb   = idx >> 8;            // ks*NBl + nbl
    int ks = fb / NBl, nbl = fb - ks * NBl;
    int hf = lane >> 4, col = lane & 15;
    int k = ks * 32 + hf * 16 + 2 * v;
    int c = nbl * 16 + col;
    float b0 = 0.f, b1 = 0.f;
    if (c < N) {
        if (bT) {
            if (k     < K) b0 = W[(size_t)c * ldb + k];
            if (k + 1 < K) b1 = W[(size_t)c * ldb + k + 1];
        } else {
            if (k     < K) b0 = W[(size_t)k * ldb + c];
            if (k + 1 < K) b1 = W[(size_t)(k + 1) * ldb + c];
        }
    }
    union { __bf16 h[2]; unsigned u; } p;
    p.h[0] = (__bf16)b0;
    p.h[1] = (__bf16)b1;
    frag[((size_t)(ks * NBtot + nbOff + nbl) * 32 + lane) * 8 + v] = p.u;
}

// 1D bias concat: o[0..n1) = a, o[n1..n1+n2) = b
__global__ void k_cat_bias(const float* __restrict__ a, int n1,
                           const float* __restrict__ b, int n2, float* __restrict__ o)
{
    int i = threadIdx.x;
    if (i < n1) o[i] = a[i];
    else if (i < n1 + n2) o[i] = b[i - n1];
}

// heads weight concat: o (64 x 16) = [mu_w (64x8) | lv_w (64x8)]
__global__ void k_cat_headw(const float* __restrict__ mu_w, const float* __restrict__ lv_w,
                            float* __restrict__ o)
{
    int idx = blockIdx.x * 256 + threadIdx.x;   // 1024
    if (idx >= 1024) return;
    int k = idx >> 4, c = idx & 15;
    o[idx] = (c < 8) ? mu_w[k * 8 + c] : lv_w[k * 8 + (c - 8)];
}

// ---------------------------------------------------------------------------
// Multi-N branch-free wave-level WMMA GEMM: one wave computes a 16 x (NB*16)
// C stripe; the A fragment is loaded ONCE per k-step and reused for all NB
// column tiles (NB compile-time so accumulators stay in VGPRs).
// C[M,N] = act( A[M,K] @ B + bias + res );  K = KS*32 exactly (A padded).
// ---------------------------------------------------------------------------
template <int NB>
__global__ void k_wmma_gemm(const float* __restrict__ A, int lda,
                            const unsigned* __restrict__ Bfrag,
                            const float* __restrict__ bias,
                            const float* __restrict__ res, int ldr,
                            float* __restrict__ C, int ldc,
                            int KS, int N, int act)
{
    const int m0 = blockIdx.x * 16;
    const int lane = threadIdx.x & 31;
    const int r  = lane & 15;
    const int hf = lane >> 4;

    v8f acc[NB];
    const v8f vz = {0.f, 0.f, 0.f, 0.f, 0.f, 0.f, 0.f, 0.f};
#pragma unroll
    for (int nb = 0; nb < NB; ++nb) acc[nb] = vz;

    const float*    ap = A + (size_t)(m0 + r) * lda + hf * 16;
    const unsigned* bp = Bfrag + lane * 8;

    for (int ks = 0; ks < KS; ++ks) {
        const float4* a4 = (const float4*)(ap + ks * 32);
        float4 a0 = a4[0], a1 = a4[1], a2 = a4[2], a3 = a4[3];
        v16bf av;
        av[0]  = (__bf16)a0.x; av[1]  = (__bf16)a0.y; av[2]  = (__bf16)a0.z; av[3]  = (__bf16)a0.w;
        av[4]  = (__bf16)a1.x; av[5]  = (__bf16)a1.y; av[6]  = (__bf16)a1.z; av[7]  = (__bf16)a1.w;
        av[8]  = (__bf16)a2.x; av[9]  = (__bf16)a2.y; av[10] = (__bf16)a2.z; av[11] = (__bf16)a2.w;
        av[12] = (__bf16)a3.x; av[13] = (__bf16)a3.y; av[14] = (__bf16)a3.z; av[15] = (__bf16)a3.w;
#pragma unroll
        for (int nb = 0; nb < NB; ++nb) {
            v16bf bv = *(const v16bf*)(bp + (size_t)(ks * NB + nb) * 256);
            acc[nb] = __builtin_amdgcn_wmma_f32_16x16x32_bf16(false, av, false, bv,
                                                              (short)0, acc[nb], false, false);
        }
    }

#pragma unroll
    for (int nb = 0; nb < NB; ++nb) {
        const int c = nb * 16 + r;
        if (c < N) {
#pragma unroll
            for (int j = 0; j < 8; ++j) {
                int row = m0 + j + 8 * hf;
                float v = acc[nb][j];
                if (bias) v += bias[c];
                if (res)  v += res[(size_t)row * ldr + c];
                if (act == 1)      v = fmaxf(v, 0.f);
                else if (act == 2) v = sigm(v);
                C[(size_t)row * ldc + c] = v;
            }
        }
    }
}

// ---------------------------------------------------------------------------
// Diagonal SSM scans over fused u buffer (N x 64: fwd cols 0-31, bwd 32-63).
// 512 fully independent scalar recurrences.
// ---------------------------------------------------------------------------
__global__ void k_ssm_scan(const float* __restrict__ u,
                           const float* __restrict__ al_f, const float* __restrict__ al_b,
                           float* __restrict__ hs_f, float* __restrict__ hs_b, int T)
{
    int tid = blockIdx.x * blockDim.x + threadIdx.x;
    int dir = tid >> 8;
    int bd  = tid & 255;
    int b = bd >> 5, d = bd & 31;
    if (dir == 0) {
        float a = sigm(al_f[d]); float h = 0.f;
        for (int t = 0; t < T; ++t) {
            size_t n = (size_t)(b * T + t);
            h = a * h + u[n * 64 + d];
            hs_f[n * 32 + d] = h;
        }
    } else {
        float a = sigm(al_b[d]); float h = 0.f;
        for (int t = T - 1; t >= 0; --t) {
            size_t n = (size_t)(b * T + t);
            h = a * h + u[n * 64 + 32 + d];
            hs_b[n * 32 + d] = h;
        }
    }
}

// ---------------------------------------------------------------------------
// GRU: one wave32 per batch; W_hh rows pinned in VGPRs, h via wave shuffles.
// gx lives inside the fused xgx buffer (stride ldg).
// ---------------------------------------------------------------------------
__global__ void k_gru_scan(const float* __restrict__ gx, int ldg,
                           const float* __restrict__ W_hh,
                           const float* __restrict__ b_hh,
                           float* __restrict__ h_seq, int T)
{
    int b = blockIdx.x;
    int d = threadIdx.x;
    float wr[32], wz[32], wn[32];
#pragma unroll
    for (int i = 0; i < 32; ++i) {
        wr[i] = W_hh[(size_t)(d)      * 32 + i];
        wz[i] = W_hh[(size_t)(32 + d) * 32 + i];
        wn[i] = W_hh[(size_t)(64 + d) * 32 + i];
    }
    float br = b_hh[d], bz = b_hh[32 + d], bn = b_hh[64 + d];
    float h = 0.f;
    for (int t = 0; t < T; ++t) {
        float gr = br, gz = bz, gn = bn;
#pragma unroll
        for (int j = 0; j < 32; ++j) {
            float hj = __shfl(h, j, 32);
            gr += wr[j] * hj; gz += wz[j] * hj; gn += wn[j] * hj;
        }
        size_t base = (size_t)(b * T + t) * ldg;
        float xr = gx[base + d], xz = gx[base + 32 + d], xn = gx[base + 64 + d];
        float rg = sigm(xr + gr);
        float zg = sigm(xz + gz);
        float nn = tanhf(xn + rg * gn);
        h = (1.f - zg) * nn + zg * h;
        h_seq[((size_t)(b * T + t)) * 32 + d] = h;
    }
}

// concat [e(256) | h_prev(32) | s_seq(32)] -> feat_in (N x 320)
__global__ void k_pack_feat(const float* __restrict__ e, const float* __restrict__ h_seq,
                            const float* __restrict__ s_seq, float* __restrict__ feat_in, int T)
{
    int n = blockIdx.x, j = threadIdx.x;   // 320 threads
    int t = n % T;
    float v;
    if (j < 256)      v = e[(size_t)n * 256 + j];
    else if (j < 288) v = (t == 0) ? 0.f : h_seq[(size_t)(n - 1) * 32 + (j - 256)];
    else              v = s_seq[(size_t)n * 32 + (j - 288)];
    feat_in[(size_t)n * 320 + j] = v;
}

// concat [e(256) | h_prev(32) | z_prev(8) | 0-pad(24)] -> sw_in (N x 320)
__global__ void k_pack_sw(const float* __restrict__ e, const float* __restrict__ h_seq,
                          const float* __restrict__ z_tilde, float* __restrict__ sw_in, int T)
{
    int n = blockIdx.x, j = threadIdx.x;   // 320 threads
    int t = n % T;
    float v;
    if (j < 256)      v = e[(size_t)n * 256 + j];
    else if (j < 288) v = (t == 0) ? 0.f : h_seq[(size_t)(n - 1) * 32 + (j - 256)];
    else if (j < 296) v = (t == 0) ? 0.f : z_tilde[(size_t)(n - 1) * 8 + (j - 288)];
    else              v = 0.f;
    sw_in[(size_t)n * 320 + j] = v;
}

// z_seq (N x 8) -> zero-padded (N x 32) so decoder GEMM is branch-free
__global__ void k_pad_z(const float* __restrict__ z, float* __restrict__ zp)
{
    int n = blockIdx.x, j = threadIdx.x;   // 32 threads
    zp[(size_t)n * 32 + j] = (j < 8) ? z[(size_t)n * 8 + j] : 0.f;
}

__global__ void k_init_kl(float* p) { *p = 0.f; }

// split heads (N x 16 = [mu|lv]) -> d_out mu/lv, z_tilde, KL reduction
__global__ void k_reparam_kl(const float* __restrict__ heads, const float* __restrict__ eps,
                             float* __restrict__ z_tilde,
                             float* __restrict__ out_mu, float* __restrict__ out_lv,
                             float* __restrict__ kl_out, int total)
{
    __shared__ float red[256];
    int i = blockIdx.x * 256 + threadIdx.x;
    float part = 0.f;
    if (i < total) {
        int n = i >> 3, j = i & 7;
        float m = heads[(size_t)n * 16 + j];
        float l = heads[(size_t)n * 16 + 8 + j];
        out_mu[i] = m;
        out_lv[i] = l;
        z_tilde[i] = m + __expf(0.5f * l) * eps[i];
        part = 1.f + l - m * m - __expf(l);
    }
    red[threadIdx.x] = part;
    __syncthreads();
    for (int s = 128; s > 0; s >>= 1) {
        if (threadIdx.x < s) red[threadIdx.x] += red[threadIdx.x + s];
        __syncthreads();
    }
    if (threadIdx.x == 0) atomicAdd(kl_out, red[0] * (-0.5f / (float)total));
}

// temporal integration: 64 independent scalar scans (b x HZ)
__global__ void k_z_scan(const float* __restrict__ beta, const float* __restrict__ z_tilde,
                         float* __restrict__ z_seq, int T)
{
    int tid = threadIdx.x;
    int b = tid >> 3, j = tid & 7;
    float z = 0.f;
    for (int t = 0; t < T; ++t) {
        size_t n = (size_t)b * T + t;
        float bt = beta[n];
        float zt = z_tilde[n * 8 + j];
        z = bt * zt + (1.f - bt) * z;
        z_seq[n * 8 + j] = z;
    }
}

// ---------------------------------------------------------------------------
// Fused hypernetwork per 16-token tile (8 waves). A_w/B_w arrive pre-packed
// as bf16 fragments (K=32 exactly, 256 column-chunks of 16). (N,4096)
// intermediates live only in registers; contractions via 16-lane butterflies.
// ---------------------------------------------------------------------------
__global__ void k_hypernet(const float* __restrict__ e,
                           const float* __restrict__ df,
                           const unsigned* __restrict__ AwF, const float* __restrict__ A_b,
                           const unsigned* __restrict__ BwF, const float* __restrict__ B_b,
                           float* __restrict__ out)
{
    __shared__ float e_tile[16][256];   // 16 KB
    __shared__ float ae[16][16];        // 1 KB
    const int m0   = blockIdx.x * 16;
    const int tid  = threadIdx.x;       // 256 = 8 waves
    const int wave = tid >> 5;
    const int lane = tid & 31;
    const int col  = lane & 15;
    const int hf   = lane >> 4;

    {   // vectorized e tile load
        float4* dst = (float4*)&e_tile[0][0];
        const float4* src = (const float4*)(e + (size_t)m0 * 256);
        for (int k = tid; k < 16 * 64; k += 256) dst[k] = src[k];
    }

    // df A-fragment (16 tokens x K=32)
    v16bf av;
    {
        const float4* dp = (const float4*)(df + (size_t)(m0 + col) * 32 + hf * 16);
        float4 a0 = dp[0], a1 = dp[1], a2 = dp[2], a3 = dp[3];
        av[0]  = (__bf16)a0.x; av[1]  = (__bf16)a0.y; av[2]  = (__bf16)a0.z; av[3]  = (__bf16)a0.w;
        av[4]  = (__bf16)a1.x; av[5]  = (__bf16)a1.y; av[6]  = (__bf16)a1.z; av[7]  = (__bf16)a1.w;
        av[8]  = (__bf16)a2.x; av[9]  = (__bf16)a2.y; av[10] = (__bf16)a2.z; av[11] = (__bf16)a2.w;
        av[12] = (__bf16)a3.x; av[13] = (__bf16)a3.y; av[14] = (__bf16)a3.z; av[15] = (__bf16)a3.w;
    }
    __syncthreads();

    // ---- A path: Ae[n,r] = ((df@A_w)+A_b)[r,:] . e[n,:] ; wave owns r, r+8
#pragma unroll
    for (int rr = 0; rr < 2; ++rr) {
        int r = wave + rr * 8;
        float part[8];
#pragma unroll
        for (int j = 0; j < 8; ++j) part[j] = 0.f;
        for (int ec = 0; ec < 16; ++ec) {
            v16bf bv = *(const v16bf*)(AwF + (size_t)(r * 16 + ec) * 256 + lane * 8);
            v8f cc = {0.f, 0.f, 0.f, 0.f, 0.f, 0.f, 0.f, 0.f};
            cc = __builtin_amdgcn_wmma_f32_16x16x32_bf16(false, av, false, bv,
                                                         (short)0, cc, false, false);
            float bias = A_b[r * 256 + ec * 16 + col];
#pragma unroll
            for (int j = 0; j < 8; ++j) {
                int row = j + 8 * hf;
                part[j] += (cc[j] + bias) * e_tile[row][ec * 16 + col];
            }
        }
#pragma unroll
        for (int j = 0; j < 8; ++j) {
            float s = part[j];
            s += __shfl_xor(s, 1, 32);
            s += __shfl_xor(s, 2, 32);
            s += __shfl_xor(s, 4, 32);
            s += __shfl_xor(s, 8, 32);
            part[j] = s;
        }
        if (col == 0) {
#pragma unroll
            for (int j = 0; j < 8; ++j) ae[j + 8 * hf][r] = part[j];
        }
    }
    __syncthreads();

    // ---- B path: out[n,e] = e[n,e] + ((df@B_w)+B_b)[e,:] . Ae[n,:]
    for (int i = 0; i < 32; ++i) {
        int ecol = wave * 32 + i;
        v16bf bv = *(const v16bf*)(BwF + (size_t)ecol * 256 + lane * 8);
        v8f cc = {0.f, 0.f, 0.f, 0.f, 0.f, 0.f, 0.f, 0.f};
        cc = __builtin_amdgcn_wmma_f32_16x16x32_bf16(false, av, false, bv,
                                                     (short)0, cc, false, false);
        float bias = B_b[ecol * 16 + col];
        float part[8];
#pragma unroll
        for (int j = 0; j < 8; ++j) {
            int row = j + 8 * hf;
            part[j] = (cc[j] + bias) * ae[row][col];
        }
#pragma unroll
        for (int j = 0; j < 8; ++j) {
            float s = part[j];
            s += __shfl_xor(s, 1, 32);
            s += __shfl_xor(s, 2, 32);
            s += __shfl_xor(s, 4, 32);
            s += __shfl_xor(s, 8, 32);
            part[j] = s;
        }
        if (col == 0) {
#pragma unroll
            for (int j = 0; j < 8; ++j) {
                int row = j + 8 * hf;
                out[(size_t)(m0 + row) * 256 + ecol] = e_tile[row][ecol] + part[j];
            }
        }
    }
}

// ---------------------------------------------------------------------------
extern "C" void kernel_launch(void* const* d_in, const int* in_sizes, int n_in,
                              void* d_out, int out_size, void* d_ws, size_t ws_size,
                              hipStream_t stream)
{
    (void)in_sizes; (void)n_in; (void)out_size; (void)ws_size;
    const int B = 8, T = 2048, E = 256, S = 32, HZ = 8;
    const int Ntok = B * T;  // 16384

    const float* e_seq = (const float*)d_in[0];
    const float* eps   = (const float*)d_in[1];
    int i = 2;
    const float* proj_in_w  = (const float*)d_in[i++];
    const float* proj_in_b  = (const float*)d_in[i++];
    const float* al_f = (const float*)d_in[i++];
    const float* Wx_f = (const float*)d_in[i++];
    const float* Wy_f = (const float*)d_in[i++];
    const float* by_f = (const float*)d_in[i++];
    const float* W1_f = (const float*)d_in[i++];
    const float* b1_f = (const float*)d_in[i++];
    const float* W2_f = (const float*)d_in[i++];
    const float* b2_f = (const float*)d_in[i++];
    const float* al_b = (const float*)d_in[i++];
    const float* Wx_b = (const float*)d_in[i++];
    const float* Wy_b = (const float*)d_in[i++];
    const float* by_b = (const float*)d_in[i++];
    const float* W1_b = (const float*)d_in[i++];
    const float* b1_b = (const float*)d_in[i++];
    const float* W2_b = (const float*)d_in[i++];
    const float* b2_b = (const float*)d_in[i++];
    const float* proj_out_w = (const float*)d_in[i++];
    const float* proj_out_b = (const float*)d_in[i++];
    const float* W_ih  = (const float*)d_in[i++];
    const float* b_ih  = (const float*)d_in[i++];
    const float* W_hh  = (const float*)d_in[i++];
    const float* b_hh  = (const float*)d_in[i++];
    const float* enc_w1 = (const float*)d_in[i++];
    const float* enc_b1 = (const float*)d_in[i++];
    const float* mu_w = (const float*)d_in[i++];
    const float* mu_b = (const float*)d_in[i++];
    const float* lv_w = (const float*)d_in[i++];
    const float* lv_b = (const float*)d_in[i++];
    const float* sw_w1 = (const float*)d_in[i++];
    const float* sw_b1 = (const float*)d_in[i++];
    const float* sw_w2 = (const float*)d_in[i++];
    const float* sw_b2 = (const float*)d_in[i++];
    const float* dec_w1 = (const float*)d_in[i++];
    const float* dec_b1 = (const float*)d_in[i++];
    const float* B_w = (const float*)d_in[i++];
    const float* B_b = (const float*)d_in[i++];
    const float* A_w = (const float*)d_in[i++];
    const float* A_b = (const float*)d_in[i++];

    float* out     = (float*)d_out;
    float* out_ec  = out;
    float* out_z   = out_ec + (size_t)Ntok * E;
    float* out_bet = out_z  + (size_t)Ntok * HZ;
    float* out_mu  = out_bet + Ntok;
    float* out_lv  = out_mu + (size_t)Ntok * HZ;
    float* out_kl  = out_lv + (size_t)Ntok * HZ;

    // ---- workspace (float granularity, offsets 32B-aligned)
    float* ws = (float*)d_ws;
    size_t off = 0;
    auto alloc = [&](size_t n) { float* p = ws + off; off += (n + 7) & ~(size_t)7; return p; };
    float* xgx     = alloc((size_t)Ntok * 128);   // [x(32) | gx(96)]
    float* u       = alloc((size_t)Ntok * 64);    // [u_f(32) | u_b(32)]
    float* hs_f    = alloc((size_t)Ntok * 32);
    float* hs_b    = alloc((size_t)Ntok * 32);
    float* h_seq   = alloc((size_t)Ntok * 32);
    float* y_f     = alloc((size_t)Ntok * 32);
    float* m_f     = alloc((size_t)Ntok * 64);
    float* yf      = alloc((size_t)Ntok * 32);
    float* y_b     = alloc((size_t)Ntok * 32);
    float* m_b     = alloc((size_t)Ntok * 64);
    float* yb      = alloc((size_t)Ntok * 32);
    float* s_seq   = alloc((size_t)Ntok * 32);
    float* feat_in = alloc((size_t)Ntok * 320);
    float* feat    = alloc((size_t)Ntok * 64);
    float* heads   = alloc((size_t)Ntok * 16);    // [mu(8) | lv(8)]
    float* z_tilde = alloc((size_t)Ntok * 8);
    float* sw_in   = alloc((size_t)Ntok * 320);
    float* sw1     = alloc((size_t)Ntok * 64);
    float* dfb     = alloc((size_t)Ntok * 32);
    float* zpad    = alloc((size_t)Ntok * 32);
    float* bias128 = alloc(128);                  // [proj_in_b | b_ih]
    float* bias16  = alloc(16);                   // [mu_b | lv_b]
    float* headw   = alloc(64 * 16);              // [mu_w | lv_w]

    auto fragsz = [](int K, int NBtot) { return (size_t)((K + 31) / 32) * NBtot * 256; };
    auto falloc = [&](int K, int NBtot) { return (unsigned*)alloc(fragsz(K, NBtot)); };
    unsigned* F_xgx   = falloc(256, 8);   // [proj_in_w | W_ih^T]
    unsigned* F_u     = falloc(32, 4);    // [Wx_f | Wx_b]
    unsigned* F_Wy_f  = falloc(32, 2);
    unsigned* F_Wy_b  = falloc(32, 2);
    unsigned* F_W1_f  = falloc(32, 4);
    unsigned* F_W1_b  = falloc(32, 4);
    unsigned* F_W2_f  = falloc(64, 2);
    unsigned* F_W2_b  = falloc(64, 2);
    unsigned* F_po_lo = falloc(32, 2);
    unsigned* F_po_hi = falloc(32, 2);
    unsigned* F_enc   = falloc(320, 4);
    unsigned* F_heads = falloc(64, 1);
    unsigned* F_sw1   = falloc(320, 4);
    unsigned* F_sw2   = falloc(64, 1);
    unsigned* F_dec   = falloc(32, 2);
    unsigned* F_Aw    = falloc(32, 256);
    unsigned* F_Bw    = falloc(32, 256);

    auto pack = [&](const float* W, int ldb, int bT, int K, int N,
                    int nbOff, int NBtot, unsigned* frag) {
        int NBl = (N + 15) / 16;
        int KS  = (K + 31) / 32;
        int total = KS * NBl * 256;
        k_pack_wfrag<<<(total + 255) / 256, 256, 0, stream>>>(W, ldb, bT, K, N, NBl,
                                                              nbOff, NBtot, total, frag);
    };
    // weight packing (tiny, once per launch; zero-pads odd K/N)
    k_cat_bias<<<1, 128, 0, stream>>>(proj_in_b, 32, b_ih, 96, bias128);
    k_cat_bias<<<1, 16, 0, stream>>>(mu_b, 8, lv_b, 8, bias16);
    k_cat_headw<<<4, 256, 0, stream>>>(mu_w, lv_w, headw);
    pack(proj_in_w, 32, 0, 256, 32, 0, 8, F_xgx);
    pack(W_ih, 256, 1, 256, 96, 2, 8, F_xgx);       // e @ W_ih^T
    pack(Wx_f, 32, 0, 32, 32, 0, 4, F_u);
    pack(Wx_b, 32, 0, 32, 32, 2, 4, F_u);
    pack(Wy_f, 32, 0, 32, 32, 0, 2, F_Wy_f);
    pack(Wy_b, 32, 0, 32, 32, 0, 2, F_Wy_b);
    pack(W1_f, 64, 0, 32, 64, 0, 4, F_W1_f);
    pack(W1_b, 64, 0, 32, 64, 0, 4, F_W1_b);
    pack(W2_f, 32, 0, 64, 32, 0, 2, F_W2_f);
    pack(W2_b, 32, 0, 64, 32, 0, 2, F_W2_b);
    pack(proj_out_w, 32, 0, 32, 32, 0, 2, F_po_lo);
    pack(proj_out_w + 32 * 32, 32, 0, 32, 32, 0, 2, F_po_hi);
    pack(enc_w1, 64, 0, 320, 64, 0, 4, F_enc);
    pack(headw, 16, 0, 64, 16, 0, 1, F_heads);
    pack(sw_w1, 64, 0, 296, 64, 0, 4, F_sw1);       // K padded 296 -> 320
    pack(sw_w2, 1, 0, 64, 1, 0, 1, F_sw2);
    pack(dec_w1, 32, 0, 8, 32, 0, 2, F_dec);        // K padded 8 -> 32
    pack(A_w, 4096, 0, 32, 4096, 0, 256, F_Aw);
    pack(B_w, 4096, 0, 32, 4096, 0, 256, F_Bw);

    auto gemm = [&](const float* A, int lda, const unsigned* Bf,
                    const float* bias, const float* res, int ldr,
                    float* C, int ldc, int K, int Ncols, int act) {
        int KS = (K + 31) / 32, NB = (Ncols + 15) / 16;
        dim3 grid(Ntok / 16);
        switch (NB) {
        case 1: k_wmma_gemm<1><<<grid, 32, 0, stream>>>(A, lda, Bf, bias, res, ldr, C, ldc, KS, Ncols, act); break;
        case 2: k_wmma_gemm<2><<<grid, 32, 0, stream>>>(A, lda, Bf, bias, res, ldr, C, ldc, KS, Ncols, act); break;
        case 4: k_wmma_gemm<4><<<grid, 32, 0, stream>>>(A, lda, Bf, bias, res, ldr, C, ldc, KS, Ncols, act); break;
        default: k_wmma_gemm<8><<<grid, 32, 0, stream>>>(A, lda, Bf, bias, res, ldr, C, ldc, KS, Ncols, act); break;
        }
    };

    // 1) fused pre-scan projections: xgx = e @ [proj_in | W_ih^T] (one A pass)
    gemm(e_seq, E, F_xgx, bias128, nullptr, 0, xgx, 128, E, 128, 0);
    gemm(xgx, 128, F_u, nullptr, nullptr, 0, u, 64, S, 64, 0);   // u = x @ [Wx_f|Wx_b]

    // 2) recurrences
    k_ssm_scan<<<2, 256, 0, stream>>>(u, al_f, al_b, hs_f, hs_b, T);
    k_gru_scan<<<B, 32, 0, stream>>>(xgx + 32, 128, W_hh, b_hh, h_seq, T);

    // 3) SSM post layers (both directions) + output projection
    gemm(hs_f, S, F_Wy_f, by_f, nullptr, 0, y_f, S, S, S, 0);
    gemm(y_f, S, F_W1_f, b1_f, nullptr, 0, m_f, 2 * S, S, 2 * S, 1);
    gemm(m_f, 2 * S, F_W2_f, b2_f, y_f, S, yf, S, 2 * S, S, 0);
    gemm(hs_b, S, F_Wy_b, by_b, nullptr, 0, y_b, S, S, S, 0);
    gemm(y_b, S, F_W1_b, b1_b, nullptr, 0, m_b, 2 * S, S, 2 * S, 1);
    gemm(m_b, 2 * S, F_W2_b, b2_b, y_b, S, yb, S, 2 * S, S, 0);
    gemm(yf, S, F_po_lo, proj_out_b, nullptr, 0, s_seq, S, S, S, 0);
    gemm(yb, S, F_po_hi, nullptr, s_seq, S, s_seq, S, S, S, 0);

    // 4) encoder + fused heads
    k_pack_feat<<<Ntok, 320, 0, stream>>>(e_seq, h_seq, s_seq, feat_in, T);
    gemm(feat_in, 320, F_enc, enc_b1, nullptr, 0, feat, 64, 320, 64, 1);
    gemm(feat, 64, F_heads, bias16, nullptr, 0, heads, 16, 64, 16, 0);

    // 5) split heads + reparameterize + KL
    k_init_kl<<<1, 1, 0, stream>>>(out_kl);
    k_reparam_kl<<<(Ntok * HZ + 255) / 256, 256, 0, stream>>>(heads, eps, z_tilde,
                                                              out_mu, out_lv, out_kl, Ntok * HZ);

    // 6) switching unit -> beta into d_out
    k_pack_sw<<<Ntok, 320, 0, stream>>>(e_seq, h_seq, z_tilde, sw_in, T);
    gemm(sw_in, 320, F_sw1, sw_b1, nullptr, 0, sw1, 64, 296, 64, 1);
    gemm(sw1, 64, F_sw2, sw_b2, nullptr, 0, out_bet, 1, 64, 1, 2);

    // 7) temporal integration -> z_seq into d_out
    k_z_scan<<<1, 64, 0, stream>>>(out_bet, z_tilde, out_z, T);

    // 8) decoder + fused hypernetwork -> e_controlled into d_out
    k_pad_z<<<Ntok, 32, 0, stream>>>(out_z, zpad);
    gemm(zpad, 32, F_dec, dec_b1, nullptr, 0, dfb, 32, HZ, 32, 1);
    k_hypernet<<<Ntok / 16, 256, 0, stream>>>(e_seq, dfb, F_Aw, A_b, F_Bw, B_b, out_ec);
}
